// Model_34986803593241
// MI455X (gfx1250) — matrix operand order repaired and checked by reading, not compiled, not run on hardware
//
#include <hip/hip_runtime.h>
#include <stdint.h>

#define DIM 10

typedef __attribute__((ext_vector_type(16))) __bf16          v16bf;
typedef __attribute__((ext_vector_type(8)))  float           v8f;
typedef __attribute__((ext_vector_type(16))) unsigned short  v16u;

// ---- bf16 helpers (round-to-nearest-even split: x ~= hi + lo) ----
__device__ __forceinline__ unsigned short f2bf(float x) {
    unsigned u = __float_as_uint(x);
    u += 0x7FFFu + ((u >> 16) & 1u);
    return (unsigned short)(u >> 16);
}
__device__ __forceinline__ float bf2f(unsigned short h) {
    return __uint_as_float(((unsigned)h) << 16);
}
__device__ __forceinline__ void atomAddF(float* p, float v) {
    // non-returning f32 atomic add, agent scope -> global_atomic_add_f32 path
    __hip_atomic_fetch_add(p, v, __ATOMIC_RELAXED, __HIP_MEMORY_SCOPE_AGENT);
}

// ---------------- zero accumulators ----------------
__global__ void zero_kernel(float* __restrict__ p, int n) {
    int i = blockIdx.x * blockDim.x + threadIdx.x;
    if (i < n) p[i] = 0.0f;
}

// ---------------- edge gather + scatter-add (the bandwidth/atomic stage) ----
// Index arrays (128 MB/etype-pair) stream through exactly once -> NT loads so
// they don't evict the L2-resident feature tables (~32 MB) and accumulators
// (~17.6 MB) that the random gathers and 176M f32 atomics depend on.
__global__ __launch_bounds__(256) void edge_scatter(
    const float* __restrict__ hsrc, const int* __restrict__ src,
    const int* __restrict__ dst, float* __restrict__ agg,
    float* __restrict__ cnt, int nE) {
    int e = blockIdx.x * blockDim.x + threadIdx.x;
    if (e >= nE) return;
    int s = __builtin_nontemporal_load(src + e);
    int d = __builtin_nontemporal_load(dst + e);
    const float2* r = (const float2*)(hsrc + (size_t)s * DIM);  // rows 8B aligned, L2-hot
    float* a = agg + (size_t)d * DIM;
#pragma unroll
    for (int k = 0; k < 5; ++k) {
        float2 v = r[k];
        atomAddF(a + 2 * k, v.x);
        atomAddF(a + 2 * k + 1, v.y);
    }
    atomAddF(cnt + d, 1.0f);
}

// ---------------- pack weights into WMMA B-operand layout (1 wave) ----------
// B layout (bf16 32x16): lane holds column N=lane%16; element e -> K = (e<8?0:16) + (lane>=16?8:0) + (e&7)
// K-blocks: [0,10) = W_hi, [10,20) = W_hi, [20,30) = W_lo, rest 0.
__global__ void pack_weights(const float* __restrict__ Wst, const float* __restrict__ Wnt,
                             const float* __restrict__ btok,
                             const float* __restrict__ Wsa, const float* __restrict__ Wna,
                             const float* __restrict__ bart,
                             const float* __restrict__ Wl, const float* __restrict__ bl,
                             unsigned short* __restrict__ Bpack, float* __restrict__ c0out) {
    int lane = threadIdx.x;
    int N = lane & 15;
    bool hiH = lane >= 16;
    for (int m = 0; m < 3; ++m) {
        unsigned short whi[DIM], wlo[DIM];
#pragma unroll
        for (int d = 0; d < DIM; ++d) {
            float w = 0.0f;
            if (N < DIM) {
                if (m == 0)      w = Wst[d * DIM + N] + Wsa[d * DIM + N];  // folded self
                else if (m == 1) w = Wnt[d * DIM + N];
                else             w = Wna[d * DIM + N];
            }
            unsigned short h = f2bf(w);
            whi[d] = h;
            wlo[d] = f2bf(w - bf2f(h));
        }
        v16u b;
        if (!hiH) {  // K = e for e<8 -> Whi[0..7]; K = 8+e for e>=8: 16..19 -> Whi[6..9], 20..23 -> Wlo[0..3]
#pragma unroll
            for (int e = 0; e < 8; ++e) b[e] = whi[e];
#pragma unroll
            for (int e = 8; e < 12; ++e) b[e] = whi[e - 2];
#pragma unroll
            for (int e = 12; e < 16; ++e) b[e] = wlo[e - 12];
        } else {     // K = 8+e for e<8: 8,9 -> Whi[8,9], 10..15 -> Whi[0..5]; K = 16+e for e>=8: 24..29 -> Wlo[4..9], 30..31 -> 0
            b[0] = whi[8]; b[1] = whi[9];
#pragma unroll
            for (int e = 2; e < 8; ++e) b[e] = whi[e - 2];
#pragma unroll
            for (int e = 8; e < 14; ++e) b[e] = wlo[e - 4];
            b[14] = 0; b[15] = 0;
        }
        *(v16u*)(Bpack + (m * 32 + lane) * 16) = b;
    }
    if (lane == 0) {  // fold biases + final bias through W_l
        float s = bl[0];
        for (int n = 0; n < DIM; ++n) s += (btok[n] + bart[n]) * Wl[n];
        *c0out = s;
    }
}

// ---------------- A-operand pack: K-blocks [hi | lo | hi] -------------------
// A layout (bf16 16x32): M = lane%16; element e -> K = (e<8?0:16) + (lane>=16?8:0) + (e&7)
__device__ __forceinline__ v16u buildA(const float x[DIM], bool hiH) {
    unsigned short hb[DIM], lb[DIM];
#pragma unroll
    for (int d = 0; d < DIM; ++d) {
        unsigned short h = f2bf(x[d]);
        hb[d] = h;
        lb[d] = f2bf(x[d] - bf2f(h));
    }
    v16u a;
    if (!hiH) {  // K = e for e<8 ; K = 8+e for e>=8
#pragma unroll
        for (int e = 0; e < 8; ++e) a[e] = hb[e];                   // K 0..7   -> hi
        a[8] = lb[6]; a[9] = lb[7]; a[10] = lb[8]; a[11] = lb[9];   // K 16..19 -> lo[6..9]
        a[12] = hb[0]; a[13] = hb[1]; a[14] = hb[2]; a[15] = hb[3]; // K 20..23 -> hi[0..3]
    } else {     // K = 8+e for e<8 ; K = 16+e for e>=8
        a[0] = hb[8]; a[1] = hb[9];                                 // K 8,9    -> hi
#pragma unroll
        for (int e = 2; e < 8; ++e) a[e] = lb[e - 2];               // K 10..15 -> lo[0..5]
#pragma unroll
        for (int e = 8; e < 14; ++e) a[e] = hb[e - 4];              // K 24..29 -> hi[4..9]
        a[14] = 0; a[15] = 0;                                       // K 30,31
    }
    return a;
}

// ---------------- per-tile WMMA stage: 1 wave = 16 schema nodes -------------
__global__ __launch_bounds__(256) void sage_tile(
    const float* __restrict__ hs,
    const float* __restrict__ agg_tok, const float* __restrict__ cnt_tok,
    const float* __restrict__ agg_art, const float* __restrict__ cnt_art,
    const unsigned short* __restrict__ Bpack, const float* __restrict__ c0p,
    const float* __restrict__ Wl,
    float* __restrict__ out, int ntiles, int nnodes) {
    int wave = (int)((blockIdx.x * blockDim.x + threadIdx.x) >> 5);
    if (wave >= ntiles) return;  // wave-uniform: EXEC stays all-ones for WMMA
    int lane = threadIdx.x & 31;
    int M = lane & 15;
    bool hiH = lane >= 16;
    int node = wave * 16 + M;

    float xr[DIM], mt[DIM], ma[DIM];
    if (node < nnodes) {
        const float2* px = (const float2*)(hs + (size_t)node * DIM);
        const float2* pt = (const float2*)(agg_tok + (size_t)node * DIM);
        const float2* pa = (const float2*)(agg_art + (size_t)node * DIM);
        float ct = fmaxf(cnt_tok[node], 1.0f);
        float ca = fmaxf(cnt_art[node], 1.0f);
#pragma unroll
        for (int k = 0; k < 5; ++k) {
            float2 v = px[k]; xr[2 * k] = v.x;      xr[2 * k + 1] = v.y;
            float2 t = pt[k]; mt[2 * k] = t.x / ct; mt[2 * k + 1] = t.y / ct;
            float2 u = pa[k]; ma[2 * k] = u.x / ca; ma[2 * k + 1] = u.y / ca;
        }
    } else {
#pragma unroll
        for (int d = 0; d < DIM; ++d) { xr[d] = 0.f; mt[d] = 0.f; ma[d] = 0.f; }
    }

    v16bf aS = __builtin_bit_cast(v16bf, buildA(xr, hiH));
    v16bf aT = __builtin_bit_cast(v16bf, buildA(mt, hiH));
    v16bf aA = __builtin_bit_cast(v16bf, buildA(ma, hiH));
    v16bf bS = __builtin_bit_cast(v16bf, *(const v16u*)(Bpack + (0 * 32 + lane) * 16));
    v16bf bT = __builtin_bit_cast(v16bf, *(const v16u*)(Bpack + (1 * 32 + lane) * 16));
    v16bf bA = __builtin_bit_cast(v16bf, *(const v16u*)(Bpack + (2 * 32 + lane) * 16));

    v8f c = {};
    c = __builtin_amdgcn_wmma_f32_16x16x32_bf16(false, aS, false, bS, (short)0, c, false, false);
    c = __builtin_amdgcn_wmma_f32_16x16x32_bf16(false, aT, false, bT, (short)0, c, false, false);
    c = __builtin_amdgcn_wmma_f32_16x16x32_bf16(false, aA, false, bA, (short)0, c, false, false);

    // C layout: element r at lane -> row M' = r + (lane>=16 ? 8 : 0), col N = lane%16.
    // Fold final Linear(10,1): per-lane scale by W_l[N], reduce across the 16-lane half.
    float wl = (M < DIM) ? Wl[M] : 0.0f;  // N == lane%16 == M here
    float t[8];
#pragma unroll
    for (int r = 0; r < 8; ++r) t[r] = c[r] * wl;
#pragma unroll
    for (int mask = 1; mask <= 8; mask <<= 1) {
#pragma unroll
        for (int r = 0; r < 8; ++r) t[r] += __shfl_xor(t[r], mask, 32);
    }
    if (M == 0) {
        float c0 = *c0p;
        int base = wave * 16 + (hiH ? 8 : 0);
#pragma unroll
        for (int r = 0; r < 8; ++r) {
            int nd = base + r;
            if (nd < nnodes) __builtin_nontemporal_store(t[r] + c0, out + nd);
        }
    }
}

extern "C" void kernel_launch(void* const* d_in, const int* in_sizes, int n_in,
                              void* d_out, int out_size, void* d_ws, size_t ws_size,
                              hipStream_t stream) {
    const float* h_schema    = (const float*)d_in[0];
    const float* h_token     = (const float*)d_in[1];
    const float* h_article   = (const float*)d_in[2];
    const int*   src_tok     = (const int*)d_in[3];
    const int*   dst_tok     = (const int*)d_in[4];
    const int*   src_art     = (const int*)d_in[5];
    const int*   dst_art     = (const int*)d_in[6];
    const float* W_self_tok  = (const float*)d_in[7];
    const float* W_neigh_tok = (const float*)d_in[8];
    const float* b_tok       = (const float*)d_in[9];
    const float* W_self_art  = (const float*)d_in[10];
    const float* W_neigh_art = (const float*)d_in[11];
    const float* b_art       = (const float*)d_in[12];
    const float* W_l         = (const float*)d_in[13];
    const float* b_l         = (const float*)d_in[14];

    int nS    = in_sizes[0] / DIM;   // 200000
    int E_tok = in_sizes[3];         // 8M
    int E_art = in_sizes[5];         // 8M

    // workspace layout (all f32 accumulators L2-resident: ~17.6 MB)
    float* ws      = (float*)d_ws;
    float* agg_tok = ws;
    float* cnt_tok = agg_tok + (size_t)nS * DIM;
    float* agg_art = cnt_tok + nS;
    float* cnt_art = agg_art + (size_t)nS * DIM;
    float* tail    = cnt_art + nS;
    uintptr_t bp = ((uintptr_t)tail + 31u) & ~(uintptr_t)31u;
    unsigned short* Bpack = (unsigned short*)bp;                       // 3*32*16 bf16 = 3 KB
    float* c0p = (float*)(bp + 3 * 32 * 16 * sizeof(unsigned short));

    int zn = nS * 22;  // 2*(10+1) floats per node
    zero_kernel<<<(zn + 255) / 256, 256, 0, stream>>>(ws, zn);
    pack_weights<<<1, 32, 0, stream>>>(W_self_tok, W_neigh_tok, b_tok,
                                       W_self_art, W_neigh_art, b_art,
                                       W_l, b_l, Bpack, c0p);
    edge_scatter<<<(E_tok + 255) / 256, 256, 0, stream>>>(h_token, src_tok, dst_tok,
                                                          agg_tok, cnt_tok, E_tok);
    edge_scatter<<<(E_art + 255) / 256, 256, 0, stream>>>(h_article, src_art, dst_art,
                                                          agg_art, cnt_art, E_art);

    int ntiles = (nS + 15) / 16;
    int nthreads = ntiles * 32;
    sage_tile<<<(nthreads + 255) / 256, 256, 0, stream>>>(h_schema, agg_tok, cnt_tok,
                                                          agg_art, cnt_art, Bpack, c0p, W_l,
                                                          (float*)d_out, ntiles, nS);
    (void)n_in; (void)out_size; (void)ws_size;
}
